// CustomGRUModel_2997887172981
// MI455X (gfx1250) — matrix operand order repaired and checked by reading, not compiled, not run on hardware
//
#include <hip/hip_runtime.h>
#include <cstddef>
#include <cstdint>

// ---------------------------------------------------------------------------
// GRU LM: logits = (GRU(emb[inputs]) @ Wd + bd)
// V=32000 E=512 U=1024 B=4 S=2048
// ---------------------------------------------------------------------------
#define GRU_V 32000
#define GRU_E 512
#define GRU_U 1024
#define GRU_B 4
#define GRU_S 2048
#define GRU_G (3 * GRU_U) // 3072

typedef __attribute__((ext_vector_type(16))) __bf16 v16bf;
typedef __attribute__((ext_vector_type(8)))  __bf16 v8bf;
typedef __attribute__((ext_vector_type(8)))  float  v8f;
typedef int gru_v4i __attribute__((vector_size(16))); // matches builtin param

// --------------------------- CDNA5 async-copy glue -------------------------
#if defined(__gfx1250__) && __has_builtin(__builtin_amdgcn_global_load_async_to_lds_b128)
#define GRU_HAS_ASYNC_LDS 1
#else
#define GRU_HAS_ASYNC_LDS 0
#endif

#if defined(__gfx1250__) && __has_builtin(__builtin_amdgcn_s_wait_asynccnt)
#define WAIT_ASYNC(n) __builtin_amdgcn_s_wait_asynccnt(n)
#elif defined(__gfx1250__)
#define WAIT_ASYNC(n) asm volatile("s_wait_asynccnt %0" ::"i"(n) : "memory")
#else
#define WAIT_ASYNC(n)
#endif

#if GRU_HAS_ASYNC_LDS
__device__ __forceinline__ void async_copy_b128(const __bf16* g, __bf16* l) {
  __builtin_amdgcn_global_load_async_to_lds_b128(
      (__attribute__((address_space(1))) gru_v4i*)(g),
      (__attribute__((address_space(3))) gru_v4i*)(l), /*offset=*/0, /*cpol=*/0);
}
#endif
// ---------------------------------------------------------------------------

__device__ __forceinline__ unsigned short f32_to_bf16(float f) {
  unsigned int u = __float_as_uint(f);
  u += 0x7FFFu + ((u >> 16) & 1u); // round-to-nearest-even
  return (unsigned short)(u >> 16);
}

__device__ __forceinline__ float fast_sigmoid(float x) {
  return 1.0f / (1.0f + __expf(-x));
}
__device__ __forceinline__ float fast_tanh(float x) {
  return 1.0f - 2.0f / (__expf(2.0f * x) + 1.0f);
}

// ---------------------------------------------------------------------------
// Embedding gather fused with fp32 -> bf16 cast.  x_bf16[b*S+s, e]
// ---------------------------------------------------------------------------
__global__ __launch_bounds__(256) void embed_cast_kernel(
    const int* __restrict__ tokens, const float* __restrict__ emb,
    unsigned short* __restrict__ xb, int total) {
  int i = blockIdx.x * blockDim.x + threadIdx.x;
  if (i >= total) return;
  int row = i >> 9;            // / E (E = 512)
  int e   = i & (GRU_E - 1);
  int tok = tokens[row];
  xb[i] = f32_to_bf16(emb[(size_t)tok * GRU_E + e]);
}

__global__ __launch_bounds__(256) void cast_bf16_kernel(
    const float* __restrict__ in, unsigned short* __restrict__ out, size_t n) {
  size_t i = (size_t)blockIdx.x * blockDim.x + threadIdx.x;
  if (i < n) out[i] = f32_to_bf16(in[i]);
}

// Wh [U, 3U] -> WhT [3U, U]   (one-time; makes GRU dot products coalesced)
__global__ __launch_bounds__(256) void transpose_wh_kernel(
    const float* __restrict__ Wh, float* __restrict__ WhT) {
  int i = blockIdx.x * blockDim.x + threadIdx.x;
  if (i >= GRU_G * GRU_U) return;
  int c = i >> 10;             // / U  : column of Wh = row of WhT
  int k = i & (GRU_U - 1);
  WhT[i] = Wh[(size_t)k * GRU_G + c];
}

__global__ __launch_bounds__(256) void zero_kernel(float* __restrict__ p, int n) {
  int i = blockIdx.x * blockDim.x + threadIdx.x;
  if (i < n) p[i] = 0.0f;
}

// ---------------------------------------------------------------------------
// Generic bf16 WMMA GEMM:  C[M,N](f32) = A[M,K](bf16,row) * B[K,N](bf16,row) + bias
// Block: 256 threads = 8 waves.  Block tile 64(M) x 128(N); wave tile 16 x 64.
// K-loop step 32 (one v_wmma_f32_16x16x32_bf16 K-slab).
// A tile double-buffered in LDS, filled with GLOBAL_LOAD_ASYNC_TO_LDS_B128
// one slab ahead (ASYNCcnt-tracked); row stride 48 elems = 96B keeps every
// 16B chunk aligned.  B fragments load straight from global (lane%16 = N-col).
// Requires: M % 64 == 0, N % 128 == 0, K % 32 == 0.
// ---------------------------------------------------------------------------
__global__ __launch_bounds__(256) void wmma_gemm_bf16_kernel(
    const unsigned short* __restrict__ Aus, const unsigned short* __restrict__ Bus,
    const float* __restrict__ bias, float* __restrict__ C,
    int M, int N, int K) {
  const __bf16* A = reinterpret_cast<const __bf16*>(Aus);
  const __bf16* B = reinterpret_cast<const __bf16*>(Bus);

  __shared__ __bf16 As[2][64 * 48]; // 2 x 6 KB

  const int tid  = threadIdx.x;
  const int wid  = tid >> 5;
  const int lane = tid & 31;
  const int wm   = wid >> 1;               // 0..3  (wave row)
  const int wn   = wid & 1;                // 0..1  (wave col)
  const int m0   = blockIdx.y * 64;
  const int n0   = blockIdx.x * 128;
  const int wm0  = m0 + wm * 16;
  const int wn0  = n0 + wn * 64;

  const int srow = tid >> 2;               // staging: row 0..63
  const int skc  = tid & 3;                // staging: 8-elem chunk 0..3
  const int half = lane >> 4;              // A frag K-interleave half
  const int mloc = wm * 16 + (lane & 15);  // A frag local row

  const __bf16* agp = A + (size_t)(m0 + srow) * K + skc * 8; // += k0 per slab
  __bf16* slot0 = &As[0][srow * 48 + skc * 8];
  __bf16* slot1 = &As[1][srow * 48 + skc * 8];

  v8f acc0 = {}, acc1 = {}, acc2 = {}, acc3 = {};

  // One K-slab: A frag from LDS (ISA 16-bit 16x32 interleave: lanes 0-15 hold
  // K 0-7/16-23, lanes 16-31 hold K 8-15/24-31), 4 B frags from global
  // (lane l = col l%16 of each 16-col tile, elems = K rows in the same interleave), 4 WMMAs.
  auto compute_tile = [&](const __bf16* abase, int i) {
    union { v16bf v; v8bf h[2]; } af;
    const __bf16* arow = abase + (size_t)mloc * 48;
    af.h[0] = *(const v8bf*)(arow + half * 8);
    af.h[1] = *(const v8bf*)(arow + 16 + half * 8);

    const int k0 = i << 5;
    const __bf16* brow = B + (size_t)k0 * N + wn0 + (lane & 15);
    v16bf b0, b1, b2, b3;
#pragma unroll
    for (int s = 0; s < 16; ++s) {
      const int kk = (s < 8) ? (half * 8 + s) : (16 + half * 8 + (s - 8));
      const __bf16* bp = brow + (size_t)kk * N;
      b0[s] = bp[0];
      b1[s] = bp[16];
      b2[s] = bp[32];
      b3[s] = bp[48];
    }

    if (k0 + 32 < K) // warm near cache for next slab's B rows
      __builtin_prefetch(B + (size_t)(k0 + 32 + lane) * N + wn0, 0, 3);

    acc0 = __builtin_amdgcn_wmma_f32_16x16x32_bf16(false, af.v, false, b0,
                                                   (short)0, acc0, false, false);
    acc1 = __builtin_amdgcn_wmma_f32_16x16x32_bf16(false, af.v, false, b1,
                                                   (short)0, acc1, false, false);
    acc2 = __builtin_amdgcn_wmma_f32_16x16x32_bf16(false, af.v, false, b2,
                                                   (short)0, acc2, false, false);
    acc3 = __builtin_amdgcn_wmma_f32_16x16x32_bf16(false, af.v, false, b3,
                                                   (short)0, acc3, false, false);
  };

  const int nT = K >> 5;

#if GRU_HAS_ASYNC_LDS
  // Software pipeline: async tile i+1 in flight while tile i is consumed.
  async_copy_b128(agp, slot0);
  for (int i = 0; i < nT; ++i) {
    if (i + 1 < nT) {
      async_copy_b128(agp + ((i + 1) << 5), ((i + 1) & 1) ? slot1 : slot0);
      WAIT_ASYNC(1); // tile i's copy (this wave) done; tile i+1 may remain
    } else {
      WAIT_ASYNC(0);
    }
    __syncthreads(); // all waves' tile-i copies visible
    compute_tile(As[i & 1], i);
    __syncthreads(); // reads done before buffer is overwritten next iter
  }
#else
  // Fallback: synchronous single-buffer staging via VGPRs.
  for (int i = 0; i < nT; ++i) {
    __syncthreads();
    *(v8bf*)slot0 = *(const v8bf*)(agp + (i << 5));
    __syncthreads();
    compute_tile(As[0], i);
  }
#endif

  // C layout: VGPR j -> M=j (lanes 0-15) / M=8+j (lanes 16-31), N = lane%16
  const int mrow = wm0 + half * 8;
  const int ncol = wn0 + (lane & 15);
  const float bi0 = bias[ncol +  0];
  const float bi1 = bias[ncol + 16];
  const float bi2 = bias[ncol + 32];
  const float bi3 = bias[ncol + 48];
#pragma unroll
  for (int j = 0; j < 8; ++j) {
    size_t r = (size_t)(mrow + j) * N + ncol;
    C[r +  0] = acc0[j] + bi0;
    C[r + 16] = acc1[j] + bi1;
    C[r + 32] = acc2[j] + bi2;
    C[r + 48] = acc3[j] + bi3;
  }
}

// ---------------------------------------------------------------------------
// One GRU timestep (Keras reset_after):
//   hg = h @ Wh + bh ; z = sig(xz+hz) ; r = sig(xr+hr) ; hc = tanh(xh + r*hh)
//   h' = z*h + (1-z)*hc
// 256 threads = 8 waves; each wave owns one hidden unit u (all 3 gates, all 4
// batches).  h (16 KB) staged in LDS; WhT rows give coalesced 128B wave reads.
// Writes h' (fp32 ping-pong) and hs[b,t,u] directly as bf16 for the decoder.
// ---------------------------------------------------------------------------
__global__ __launch_bounds__(256) void gru_step_kernel(
    const float* __restrict__ h_in, float* __restrict__ h_out,
    const float* __restrict__ xg, const float* __restrict__ WhT,
    const float* __restrict__ bh, unsigned short* __restrict__ hs_bf16, int t) {
  __shared__ float hsh[GRU_B * GRU_U]; // 16 KB

  const int tid = threadIdx.x;
  for (int i = tid; i < GRU_B * GRU_U; i += 256) hsh[i] = h_in[i];
  __syncthreads();

  const int wid  = tid >> 5;
  const int lane = tid & 31;
  const int u    = blockIdx.x * 8 + wid;

  const float* wz = WhT + (size_t)u * GRU_U;
  const float* wr = WhT + (size_t)(GRU_U + u) * GRU_U;
  const float* wh = WhT + (size_t)(2 * GRU_U + u) * GRU_U;

  float az[GRU_B] = {0.f, 0.f, 0.f, 0.f};
  float ar[GRU_B] = {0.f, 0.f, 0.f, 0.f};
  float ah[GRU_B] = {0.f, 0.f, 0.f, 0.f};

  for (int k = lane; k < GRU_U; k += 32) {
    const float wzv = wz[k], wrv = wr[k], whv = wh[k];
#pragma unroll
    for (int b = 0; b < GRU_B; ++b) {
      const float hv = hsh[b * GRU_U + k];
      az[b] = __fmaf_rn(hv, wzv, az[b]);
      ar[b] = __fmaf_rn(hv, wrv, ar[b]);
      ah[b] = __fmaf_rn(hv, whv, ah[b]);
    }
  }

#pragma unroll
  for (int off = 16; off > 0; off >>= 1) {
#pragma unroll
    for (int b = 0; b < GRU_B; ++b) {
      az[b] += __shfl_down(az[b], off, 32);
      ar[b] += __shfl_down(ar[b], off, 32);
      ah[b] += __shfl_down(ah[b], off, 32);
    }
  }

  if (lane == 0) {
    const float bz = bh[u], br = bh[GRU_U + u], bb = bh[2 * GRU_U + u];
#pragma unroll
    for (int b = 0; b < GRU_B; ++b) {
      const float* x = xg + ((size_t)b * GRU_S + t) * GRU_G;
      const float z  = fast_sigmoid(x[u] + az[b] + bz);
      const float r  = fast_sigmoid(x[GRU_U + u] + ar[b] + br);
      const float hc = fast_tanh(x[2 * GRU_U + u] + r * (ah[b] + bb));
      const float hp = hsh[b * GRU_U + u];
      const float hn = z * hp + (1.0f - z) * hc;
      h_out[b * GRU_U + u] = hn;
      hs_bf16[((size_t)b * GRU_S + t) * GRU_U + u] = f32_to_bf16(hn);
    }
  }
}

// ---------------------------------------------------------------------------
// Host orchestration
// ---------------------------------------------------------------------------
extern "C" void kernel_launch(void* const* d_in, const int* in_sizes, int n_in,
                              void* d_out, int out_size, void* d_ws, size_t ws_size,
                              hipStream_t stream) {
  const int*   tokens = (const int*)d_in[0];
  const float* emb    = (const float*)d_in[1];
  const float* Wx     = (const float*)d_in[2];
  const float* Wh     = (const float*)d_in[3];
  const float* bx     = (const float*)d_in[4];
  const float* bh     = (const float*)d_in[5];
  const float* Wd     = (const float*)d_in[6];
  const float* bd     = (const float*)d_in[7];
  float* out = (float*)d_out;

  char* ws = (char*)d_ws;
  size_t off = 0;
  auto alloc = [&](size_t bytes) -> void* {
    void* p = ws + off;
    off += (bytes + 255) & ~(size_t)255;
    return p;
  };

  unsigned short* xb  = (unsigned short*)alloc((size_t)GRU_B * GRU_S * GRU_E * 2);
  unsigned short* Wxb = (unsigned short*)alloc((size_t)GRU_E * GRU_G * 2);
  unsigned short* Wdb = (unsigned short*)alloc((size_t)GRU_U * GRU_V * 2);
  float*          WhT = (float*)alloc((size_t)GRU_G * GRU_U * 4);
  float*          xg  = (float*)alloc((size_t)GRU_B * GRU_S * GRU_G * 4);
  unsigned short* hsb = (unsigned short*)alloc((size_t)GRU_B * GRU_S * GRU_U * 2);
  float*          h0  = (float*)alloc((size_t)GRU_B * GRU_U * 4);
  float*          h1  = (float*)alloc((size_t)GRU_B * GRU_U * 4);

  // 1. embedding gather (+ bf16 cast) and weight prep
  {
    int total = GRU_B * GRU_S * GRU_E;
    embed_cast_kernel<<<(total + 255) / 256, 256, 0, stream>>>(tokens, emb, xb, total);
  }
  {
    size_t n = (size_t)GRU_E * GRU_G;
    cast_bf16_kernel<<<(unsigned)((n + 255) / 256), 256, 0, stream>>>(Wx, Wxb, n);
  }
  {
    size_t n = (size_t)GRU_U * GRU_V;
    cast_bf16_kernel<<<(unsigned)((n + 255) / 256), 256, 0, stream>>>(Wd, Wdb, n);
  }
  {
    int n = GRU_G * GRU_U;
    transpose_wh_kernel<<<(n + 255) / 256, 256, 0, stream>>>(Wh, WhT);
  }

  // 2. xg = x @ Wx + bx   [8192,512] x [512,3072]
  {
    dim3 grid(GRU_G / 128, (GRU_B * GRU_S) / 64);
    wmma_gemm_bf16_kernel<<<grid, 256, 0, stream>>>(xb, Wxb, bx, xg,
                                                    GRU_B * GRU_S, GRU_G, GRU_E);
  }

  // 3. sequential GRU scan (2048 graph-captured step launches)
  zero_kernel<<<(GRU_B * GRU_U + 255) / 256, 256, 0, stream>>>(h0, GRU_B * GRU_U);
  float* hc = h0;
  float* hn = h1;
  for (int t = 0; t < GRU_S; ++t) {
    gru_step_kernel<<<GRU_U / 8, 256, 0, stream>>>(hc, hn, xg, WhT, bh, hsb, t);
    float* tmp = hc; hc = hn; hn = tmp;
  }

  // 4. logits = hs @ Wd + bd   [8192,1024] x [1024,32000]
  {
    dim3 grid(GRU_V / 128, (GRU_B * GRU_S) / 64);
    wmma_gemm_bf16_kernel<<<grid, 256, 0, stream>>>(hsb, Wdb, bd, out,
                                                    GRU_B * GRU_S, GRU_V, GRU_U);
  }
}
